// WordProbLayer_57466662420793
// MI455X (gfx1250) — matrix operand order repaired and checked by reading, not compiled, not run on hardware
//
#include <hip/hip_runtime.h>
#include <hip/hip_bf16.h>
#include <math.h>

// ---------------- problem constants (fixed by the reference) ----------------
#define T_   32
#define B_   32
#define S_   400
#define H_   512
#define V_   50000
#define EXT_ 64
#define TB_  (T_ * B_)          // 1024 rows
#define F3H  (3 * H_)           // 1536
#define LDP  (V_ + EXT_)        // 50064  (pred row stride in d_out)

// ---------------- WMMA types ----------------
typedef __attribute__((ext_vector_type(16))) _Float16 v16h;
typedef __attribute__((ext_vector_type(8)))  _Float16 v8h;
typedef __attribute__((ext_vector_type(4)))  _Float16 v4h;
typedef __attribute__((ext_vector_type(8)))  float    v8f;

// ---------------- GEMM tiling ----------------
#define BM 128
#define BN 128
#define BK 32
#define SA 40   // padded LDS row stride in f16 elems (80B, 16B-aligned)

#define USE_ASYNC_LDS 1

// copy 16 bytes global(f16) -> LDS; async path uses the CDNA5 ASYNCcnt DMA
__device__ __forceinline__ void copy16_to_lds(_Float16* lds_ptr, const _Float16* gptr)
{
#if USE_ASYNC_LDS
    // low 32 bits of a flat pointer into the LDS aperture = workgroup-relative
    // LDS byte offset
    unsigned lds_off = (unsigned)(uintptr_t)lds_ptr;
    asm volatile("global_load_async_to_lds_b128 %0, %1, off"
                 :: "v"(lds_off), "v"(gptr) : "memory");
#else
    *(v8h*)lds_ptr = *(const v8h*)gptr;
#endif
}

__device__ __forceinline__ void wait_async_tail() {
#if USE_ASYNC_LDS
    asm volatile("s_wait_asynccnt 0x4" ::: "memory");   // next stage (4 ops) may fly
#endif
}
__device__ __forceinline__ void wait_async_all() {
#if USE_ASYNC_LDS
    asm volatile("s_wait_asynccnt 0x0" ::: "memory");
#endif
}

// ---------------- f32 -> f16 conversion (n must be a multiple of 4) ----------------
__global__ __launch_bounds__(256)
void cvt_f32_f16(const float* __restrict__ in, _Float16* __restrict__ out, long n)
{
    long i = ((long)blockIdx.x * 256 + threadIdx.x) * 4;
    if (i < n) {
        float4 v = *(const float4*)(in + i);
        v4h o;
        o[0] = (_Float16)v.x; o[1] = (_Float16)v.y;
        o[2] = (_Float16)v.z; o[3] = (_Float16)v.w;
        *(v4h*)(out + i) = o;
    }
}

// Y[m,n] = scale * ( sum_k X[m,k] * W[n,k] + bias[n] ),  X:[M,K] f16, W:[N,K] f16.
// Block tile 128x128, 8 waves, wave tile 32x64 (2x4 WMMA tiles).
// Requires M % 128 == 0, K % 32 == 0 (true at all call sites); N guarded.
// Double-buffered LDS pipeline fed by async global->LDS copies (ASYNCcnt).
__global__ __launch_bounds__(256)
void gemm_nt_wmma_f16(const _Float16* __restrict__ X, const _Float16* __restrict__ W,
                      const float* __restrict__ bias, float* __restrict__ Y,
                      int M, int N, int K, long ldy, float scale)
{
    __shared__ __align__(16) _Float16 As[2][BM * SA];
    __shared__ __align__(16) _Float16 Bs[2][BN * SA];

    const int tid  = threadIdx.x;
    const int lane = tid & 31;
    const int w    = tid >> 5;
    const int wm   = (w & 3) * 32;    // wave M offset inside block tile
    const int wn   = (w >> 2) * 64;   // wave N offset inside block tile
    const long m0  = (long)blockIdx.y * BM;
    const long n0  = (long)blockIdx.x * BN;

    // fragment lane mapping (CDNA5 16-bit WMMA layouts)
    const int r  = lane & 15;
    const int hv = lane >> 4;

    // staging assignments (16B chunks; 512 chunks per 128x32 tile, 2 per thread)
    const int cr0 = (tid      ) >> 2, cc0 = ((tid      ) & 3) << 3;
    const int cr1 = (tid + 256) >> 2, cc1 = ((tid + 256) & 3) << 3;
    long bn0 = n0 + cr0; if (bn0 >= N) bn0 = N - 1;   // clamp OOB W rows
    long bn1 = n0 + cr1; if (bn1 >= N) bn1 = N - 1;

    v8f acc[2][4] = {};

    const int nk = K / BK;
    auto stage = [&](int buf, int k0) {
        copy16_to_lds(&As[buf][cr0 * SA + cc0], X + (m0 + cr0) * (long)K + k0 + cc0);
        copy16_to_lds(&As[buf][cr1 * SA + cc1], X + (m0 + cr1) * (long)K + k0 + cc1);
        copy16_to_lds(&Bs[buf][cr0 * SA + cc0], W + bn0 * (long)K + k0 + cc0);
        copy16_to_lds(&Bs[buf][cr1 * SA + cc1], W + bn1 * (long)K + k0 + cc1);
    };

    stage(0, 0);
    if (nk > 1) stage(1, BK);

    for (int i = 0; i < nk; ++i) {
        if (i + 1 < nk) wait_async_tail();   // stage i done; stage i+1 may be in flight
        else            wait_async_all();
        __syncthreads();

        const int buf = i & 1;
        #pragma unroll
        for (int im = 0; im < 2; ++im) {
            // A frag 16x32: lane (r,hv): elems 0..7 -> K=hv*8+j, 8..15 -> K=16+hv*8+j
            const _Float16* ap = &As[buf][(wm + im * 16 + r) * SA];
            v8h alo = *(const v8h*)(ap + hv * 8);
            v8h ahi = *(const v8h*)(ap + 16 + hv * 8);
            v16h a;
            #pragma unroll
            for (int j = 0; j < 8; ++j) { a[j] = alo[j]; a[j + 8] = ahi[j]; }
            #pragma unroll
            for (int jn = 0; jn < 4; ++jn) {
                // B frag 32x16: lane (n=r, half hv): elems 0..7 -> K=hv*8+j, 8..15 -> K=16+hv*8+j
                const _Float16* bp = &Bs[buf][(wn + jn * 16 + r) * SA];
                v8h blo = *(const v8h*)(bp + hv * 8);
                v8h bhi = *(const v8h*)(bp + 16 + hv * 8);
                v16h bfrag;
                #pragma unroll
                for (int j = 0; j < 8; ++j) { bfrag[j] = blo[j]; bfrag[j + 8] = bhi[j]; }
                acc[im][jn] = __builtin_amdgcn_wmma_f32_16x16x32_f16(
                    false, a, false, bfrag, (short)0, acc[im][jn], false, false);
            }
        }
        __syncthreads();                      // everyone done reading buf
        if (i + 2 < nk) stage(buf, (i + 2) * BK);
        if (i + 3 < nk)
            __builtin_prefetch(X + (m0 + (tid >> 1)) * (long)K + (i + 3) * BK, 0, 1);
    }

    // ---- epilogue: D layout: VGPR e -> row = hv*8 + e, col = r ----
    #pragma unroll
    for (int im = 0; im < 2; ++im) {
        #pragma unroll
        for (int jn = 0; jn < 4; ++jn) {
            long col = n0 + wn + jn * 16 + r;
            if (col < N) {
                float bvv = bias ? bias[col] : 0.0f;
                #pragma unroll
                for (int e = 0; e < 8; ++e) {
                    long row = m0 + wm + im * 16 + hv * 8 + e;
                    Y[row * ldy + col] = scale * (acc[im][jn][e] + bvv);
                }
            }
        }
    }
}

// ---------------- wave32 reductions ----------------
__device__ __forceinline__ float warp_sum(float v) {
    #pragma unroll
    for (int o = 16; o > 0; o >>= 1) v += __shfl_xor(v, o, 32);
    return v;
}
__device__ __forceinline__ float warp_max(float v) {
    #pragma unroll
    for (int o = 16; o > 0; o >>= 1) v = fmaxf(v, __shfl_xor(v, o, 32));
    return v;
}

// scores[t,b,s] = q[t,b,:] . k[s,b,:]   (q already scaled), with key-padding mask
__global__ __launch_bounds__(256)
void scores_kernel(const float* __restrict__ q, const float* __restrict__ k,
                   const unsigned char* __restrict__ mask, float* __restrict__ scores)
{
    int tb = blockIdx.x;                 // t*B + b
    int b  = tb % B_;
    __shared__ float qs[H_];
    for (int d = threadIdx.x; d < H_; d += blockDim.x) qs[d] = q[(long)tb * H_ + d];
    __syncthreads();
    for (int s = threadIdx.x; s < S_; s += blockDim.x) {
        const float* kr = k + ((long)s * B_ + b) * H_;
        float acc = 0.f;
        #pragma unroll 8
        for (int d = 0; d < H_; ++d) acc += qs[d] * kr[d];
        scores[(long)tb * S_ + s] = mask[b * S_ + s] ? -1e9f : acc;
    }
}

// row softmax over S=400; writes dists (final output #2)
__global__ __launch_bounds__(128)
void softmax_s_kernel(const float* __restrict__ scores, float* __restrict__ dists)
{
    __shared__ float red[4];
    int row = blockIdx.x;
    const float* x = scores + (long)row * S_;
    float* y = dists + (long)row * S_;
    int lane = threadIdx.x & 31, wv = threadIdx.x >> 5;

    float m = -1e30f;
    for (int s = threadIdx.x; s < S_; s += 128) m = fmaxf(m, x[s]);
    m = warp_max(m);
    if (lane == 0) red[wv] = m;
    __syncthreads();
    m = fmaxf(fmaxf(red[0], red[1]), fmaxf(red[2], red[3]));

    float sum = 0.f;
    for (int s = threadIdx.x; s < S_; s += 128) sum += __expf(x[s] - m);
    sum = warp_sum(sum);
    __syncthreads();
    if (lane == 0) red[wv] = sum;
    __syncthreads();
    float inv = 1.f / (red[0] + red[1] + red[2] + red[3]);
    for (int s = threadIdx.x; s < S_; s += 128) y[s] = __expf(x[s] - m) * inv;
}

// ctx[t,b,d] = sum_s dists[t,b,s] * v[s,b,d]
__global__ __launch_bounds__(256)
void ctx_kernel(const float* __restrict__ dists, const float* __restrict__ v,
                float* __restrict__ ctx)
{
    int tb = blockIdx.x;
    int b  = tb % B_;
    __shared__ float dw[S_];
    for (int s = threadIdx.x; s < S_; s += blockDim.x) dw[s] = dists[(long)tb * S_ + s];
    __syncthreads();
    for (int d = threadIdx.x; d < H_; d += blockDim.x) {
        float acc = 0.f;
        for (int s = 0; s < S_; ++s) acc += dw[s] * v[((long)s * B_ + b) * H_ + d];
        ctx[(long)tb * H_ + d] = acc;
    }
}

// feat[:, 0:H] = h ; feat[:, H:2H] = y_emb   (cols 2H:3H written by Wo GEMM)
__global__ __launch_bounds__(256)
void concat_kernel(const float* __restrict__ h, const float* __restrict__ ye,
                   float* __restrict__ feat)
{
    long i = (long)blockIdx.x * 256 + threadIdx.x;
    if (i >= (long)TB_ * H_) return;
    long m = i / H_;
    int  d = (int)(i % H_);
    feat[m * F3H + d]      = h[i];
    feat[m * F3H + H_ + d] = ye[i];
}

// g[m] = sigmoid(feat[m,:] . Wcopy + bcopy)
__global__ __launch_bounds__(128)
void copygate_kernel(const float* __restrict__ feat, const float* __restrict__ Wc,
                     const float* __restrict__ bc, float* __restrict__ g)
{
    __shared__ float red[4];
    int row = blockIdx.x;
    const float* f = feat + (long)row * F3H;
    float acc = 0.f;
    for (int d = threadIdx.x; d < F3H; d += 128) acc += f[d] * Wc[d];
    acc = warp_sum(acc);
    int lane = threadIdx.x & 31, wv = threadIdx.x >> 5;
    if (lane == 0) red[wv] = acc;
    __syncthreads();
    if (threadIdx.x == 0) {
        float s = red[0] + red[1] + red[2] + red[3] + bc[0];
        g[row] = 1.f / (1.f + __expf(-s));
    }
}

// in-place row softmax over V (row stride LDP), scaled by g, zero the EXT tail
__global__ __launch_bounds__(256)
void softmax_vocab_kernel(float* __restrict__ pred, const float* __restrict__ g)
{
    __shared__ float red[8];
    int row = blockIdx.x;
    float* x = pred + (long)row * LDP;
    int lane = threadIdx.x & 31, wv = threadIdx.x >> 5;

    float m = -1e30f;
    for (int n = threadIdx.x; n < V_; n += 256) m = fmaxf(m, x[n]);
    m = warp_max(m);
    if (lane == 0) red[wv] = m;
    __syncthreads();
    float mm = red[0];
    #pragma unroll
    for (int i = 1; i < 8; ++i) mm = fmaxf(mm, red[i]);

    float sum = 0.f;
    for (int n = threadIdx.x; n < V_; n += 256) sum += __expf(x[n] - mm);
    sum = warp_sum(sum);
    __syncthreads();
    if (lane == 0) red[wv] = sum;
    __syncthreads();
    float ss = 0.f;
    #pragma unroll
    for (int i = 0; i < 8; ++i) ss += red[i];

    float sc = g[row] / ss;
    for (int n = threadIdx.x; n < V_; n += 256) x[n] = __expf(x[n] - mm) * sc;
    for (int n = V_ + threadIdx.x; n < LDP; n += 256) x[n] = 0.f;
}

// pred[t,b, xids[s,b]] += (1-g[t,b]) * dists[t,b,s]
__global__ __launch_bounds__(256)
void scatter_kernel(const float* __restrict__ dists, const float* __restrict__ g,
                    const int* __restrict__ xids, float* __restrict__ pred)
{
    long i = (long)blockIdx.x * 256 + threadIdx.x;
    if (i >= (long)TB_ * S_) return;
    int  s  = (int)(i % S_);
    long tb = i / S_;
    int  b  = (int)(tb % B_);
    int  idx = xids[(long)s * B_ + b];
    float w = (1.f - g[tb]) * dists[i];
    atomicAdd(pred + tb * (long)LDP + idx, w);
}

extern "C" void kernel_launch(void* const* d_in, const int* in_sizes, int n_in,
                              void* d_out, int out_size, void* d_ws, size_t ws_size,
                              hipStream_t stream)
{
    (void)in_sizes; (void)n_in; (void)out_size; (void)ws_size;
    const float* h      = (const float*)d_in[0];
    const float* y_emb  = (const float*)d_in[1];
    const float* memory = (const float*)d_in[2];
    const unsigned char* mask = (const unsigned char*)d_in[3];
    const int*   xids   = (const int*)d_in[4];
    // d_in[5] = max_ext_len (compile-time EXT_)
    const float* Wq = (const float*)d_in[6],  *bq = (const float*)d_in[7];
    const float* Wk = (const float*)d_in[8],  *bk = (const float*)d_in[9];
    const float* Wv = (const float*)d_in[10], *bv = (const float*)d_in[11];
    const float* Wo = (const float*)d_in[12], *bo = (const float*)d_in[13];
    const float* Wproj = (const float*)d_in[14], *bproj = (const float*)d_in[15];
    const float* Wcopy = (const float*)d_in[16], *bcopy = (const float*)d_in[17];

    // ---- workspace layout: f32 region then f16 region (~240 MB total) ----
    float* ws     = (float*)d_ws;
    float* q      = ws;                              // TB_*H_
    float* kbuf   = q      + (long)TB_ * H_;         // S_*B_*H_
    float* vbuf   = kbuf   + (long)S_ * B_ * H_;     // S_*B_*H_
    float* scores = vbuf   + (long)S_ * B_ * H_;     // TB_*S_
    float* ctx    = scores + (long)TB_ * S_;         // TB_*H_
    float* feat   = ctx    + (long)TB_ * H_;         // TB_*F3H
    float* g      = feat   + (long)TB_ * F3H;        // TB_

    _Float16* h16     = (_Float16*)(g + TB_);
    _Float16* mem16   = h16    + (long)TB_ * H_;
    _Float16* ctx16   = mem16  + (long)S_ * B_ * H_;
    _Float16* feat16  = ctx16  + (long)TB_ * H_;
    _Float16* wq16    = feat16 + (long)TB_ * F3H;
    _Float16* wk16    = wq16   + (long)H_ * H_;
    _Float16* wv16    = wk16   + (long)H_ * H_;
    _Float16* wo16    = wv16   + (long)H_ * H_;
    _Float16* wproj16 = wo16   + (long)H_ * H_;      // V_*F3H (153 MB, fits L2 at f16)

    float* pred   = (float*)d_out;                   // TB_ * LDP
    float* dists  = pred + (long)TB_ * LDP;          // TB_ * S_

    const float qscale = 0.044194173824159216f;      // 1/sqrt(512)
    dim3 blk(256);
    auto cvt = [&](const float* src, _Float16* dst, long n) {
        cvt_f32_f16<<<(unsigned)((n / 4 + 255) / 256), blk, 0, stream>>>(src, dst, n);
    };

    // one-time f16 conversions (bandwidth-trivial; halves GEMM traffic)
    cvt(h,      h16,    (long)TB_ * H_);
    cvt(memory, mem16,  (long)S_ * B_ * H_);
    cvt(Wq,     wq16,   (long)H_ * H_);
    cvt(Wk,     wk16,   (long)H_ * H_);
    cvt(Wv,     wv16,   (long)H_ * H_);
    cvt(Wo,     wo16,   (long)H_ * H_);
    cvt(Wproj,  wproj16,(long)V_ * F3H);

    // q/k/v projections (WMMA)
    gemm_nt_wmma_f16<<<dim3(H_ / BN, TB_ / BM), blk, 0, stream>>>(
        h16, wq16, bq, q, TB_, H_, H_, H_, qscale);
    gemm_nt_wmma_f16<<<dim3(H_ / BN, (S_ * B_) / BM), blk, 0, stream>>>(
        mem16, wk16, bk, kbuf, S_ * B_, H_, H_, H_, 1.0f);
    gemm_nt_wmma_f16<<<dim3(H_ / BN, (S_ * B_) / BM), blk, 0, stream>>>(
        mem16, wv16, bv, vbuf, S_ * B_, H_, H_, H_, 1.0f);

    // attention
    scores_kernel<<<TB_, 256, 0, stream>>>(q, kbuf, mask, scores);
    softmax_s_kernel<<<TB_, 128, 0, stream>>>(scores, dists);
    ctx_kernel<<<TB_, 256, 0, stream>>>(dists, vbuf, ctx);

    // atts = ctx @ Wo.T + bo, written into feat[:, 2H:3H]
    cvt(ctx, ctx16, (long)TB_ * H_);
    gemm_nt_wmma_f16<<<dim3(H_ / BN, TB_ / BM), blk, 0, stream>>>(
        ctx16, wo16, bo, feat + 2 * H_, TB_, H_, H_, (long)F3H, 1.0f);
    concat_kernel<<<(TB_ * H_ + 255) / 256, 256, 0, stream>>>(h, y_emb, feat);

    // copy gate
    copygate_kernel<<<TB_, 128, 0, stream>>>(feat, Wcopy, bcopy, g);

    // vocab logits straight into d_out with row stride LDP (157 GFLOP WMMA)
    cvt(feat, feat16, (long)TB_ * F3H);
    gemm_nt_wmma_f16<<<dim3((V_ + BN - 1) / BN, TB_ / BM), blk, 0, stream>>>(
        feat16, wproj16, bproj, pred, TB_, V_, F3H, (long)LDP, 1.0f);

    // softmax * g (in place) + zero EXT tail, then scatter-add copy distribution
    softmax_vocab_kernel<<<TB_, 256, 0, stream>>>(pred, g);
    scatter_kernel<<<(TB_ * S_ + 255) / 256, 256, 0, stream>>>(dists, g, xids, pred);
}
